// NTM_CopyTask_Model_16346645529315
// MI455X (gfx1250) — matrix operand order repaired and checked by reading, not compiled, not run on hardware
//
#include <hip/hip_runtime.h>
#include <hip/hip_bf16.h>

// ---------------------------------------------------------------------------
// NTM copy-task forward for MI455X (gfx1250, wave32, WMMA bf16 16x16x32)
// ---------------------------------------------------------------------------
// B=64, T=64, D=64, H=1024, MR=128, MC=64, R=1, Wr=1, NH=2, EPS=1e-8
// K_lstm = 1152 ; head_dim = 268 (pad 272) ; K_out = 1088 ; 126 steps
// LSTM kernel: one wave per (16-row, 16-col) tile carrying FOUR accumulators
// (i,f,g,o) -> A fragment loaded once per K-step, 4 WMMAs, gates fused in
// registers (no LDS, no barriers).
// ---------------------------------------------------------------------------

typedef __attribute__((ext_vector_type(16))) __bf16 v16bf;
typedef __attribute__((ext_vector_type(8)))  __bf16 v8bf;
typedef __attribute__((ext_vector_type(8)))  float  v8f;

#define EPSC 1e-8f

__device__ __forceinline__ float sigf(float x)      { return 1.0f / (1.0f + expf(-x)); }
__device__ __forceinline__ float softplusf(float x) { return (x > 20.0f) ? x : log1pf(expf(x)); }

// A-fragment (16x32 bf16): per-lane two contiguous 16B loads, K k0..k0+7 in
// VGPRs 0-3 and K k0+8..k0+15 in VGPRs 4-7 (k0 already folded into Arow).
__device__ __forceinline__ v16bf load_afrag(const __bf16* __restrict__ Arow, int kt) {
    const __bf16* ap = Arow + kt * 32;
    v8bf alo = *(const v8bf*)(ap);
    v8bf ahi = *(const v8bf*)(ap + 8);
    return __builtin_shufflevector(alo, ahi,
                                   0, 1, 2, 3, 4, 5, 6, 7,
                                   8, 9, 10, 11, 12, 13, 14, 15);
}

// B-fragment (32x16 bf16, column-major weights): 16 consecutive K per lane.
__device__ __forceinline__ v8f wmma_mac_a(v16bf a, const __bf16* __restrict__ Brow,
                                          int kt, v8f acc) {
    v16bf b = *(const v16bf*)(Brow + kt * 32);
    return __builtin_amdgcn_wmma_f32_16x16x32_bf16(
        /*neg_a=*/false, a, /*neg_b=*/false, b,
        /*c_mod=*/(short)0, acc, /*reuse_a=*/false, /*reuse_b=*/false);
}

__device__ __forceinline__ v8f wmma_mac(const __bf16* __restrict__ Arow,
                                        const __bf16* __restrict__ Brow,
                                        int kt, v8f acc) {
    return wmma_mac_a(load_afrag(Arow, kt), Brow, kt, acc);
}

// ---------------------------------------------------------------------------
// Prep: f32 (K x N) -> bf16 column-major (Npad x K), zero pad columns.
// ---------------------------------------------------------------------------
__global__ void k_cvt_t(const float* __restrict__ W, __bf16* __restrict__ Wt,
                        int K, int N, int Npad) {
    int idx = blockIdx.x * blockDim.x + threadIdx.x;
    int total = Npad * K;
    if (idx >= total) return;
    int col = idx / K;
    int k   = idx - col * K;
    float v = (col < N) ? W[(size_t)k * N + col] : 0.0f;
    Wt[idx] = (__bf16)v;
}

__global__ void k_fill_f32(float* __restrict__ p, float v, int n) {
    int i = blockIdx.x * blockDim.x + threadIdx.x;
    if (i < n) p[i] = v;
}
__global__ void k_fill_bf16(__bf16* __restrict__ p, int n) {
    int i = blockIdx.x * blockDim.x + threadIdx.x;
    if (i < n) p[i] = (__bf16)0.0f;
}
__global__ void k_init_wprev(float* __restrict__ w, int n) {
    int i = blockIdx.x * blockDim.x + threadIdx.x;
    if (i < n) w[i] = ((i & 127) == 0) ? 1.0f : 0.0f;   // w0[:, :, 0] = 1
}

// ---------------------------------------------------------------------------
// Per-step: build bf16 activation matrix A = [x_t | r_prev | h]  (64 x 1152)
// ---------------------------------------------------------------------------
__global__ void k_pack(__bf16* __restrict__ A, const float* __restrict__ x,
                       const __bf16* __restrict__ rbf,
                       const __bf16* __restrict__ hbf, int t) {
    int b = blockIdx.x;
    for (int j = threadIdx.x; j < 1152; j += blockDim.x) {
        __bf16 v;
        if (j < 64)       v = (t < 64) ? (__bf16)x[(b * 64 + t) * 64 + j] : (__bf16)0.0f;
        else if (j < 128) v = rbf[b * 64 + (j - 64)];
        else              v = hbf[b * 1024 + (j - 128)];
        A[b * 1152 + j] = v;
    }
}

// ---------------------------------------------------------------------------
// LSTM GEMM + fused gates.  grid = (H/16=64, B/16=4), block = 32 (1 wave).
// One wave computes the 16x16 tile for ALL FOUR gates: 4 accumulators,
// A fragment loaded once per K-step, gates fused in registers.
// ---------------------------------------------------------------------------
__global__ void k_lstm(const __bf16* __restrict__ A,
                       const __bf16* __restrict__ WtL,
                       const float* __restrict__ b_lstm,
                       float* __restrict__ c,
                       __bf16* __restrict__ hbf,
                       __bf16* __restrict__ hr) {
    const int lane = threadIdx.x & 31;
    const int ct = blockIdx.x;                // hidden-col tile (H/16)
    const int rt = blockIdx.y;                // batch-row tile (B/16)
    const int lhalf = lane >> 4;
    const int lmod  = lane & 15;

    const int colH = ct * 16 + lmod;          // hidden unit column (per lane)
    const __bf16* Arow = A + (size_t)(rt * 16 + lmod) * 1152 + (lhalf ? 16 : 0);
    const __bf16* Brow0 = WtL + (size_t)(0 * 1024 + colH) * 1152 + (lhalf ? 16 : 0);
    const __bf16* Brow1 = WtL + (size_t)(1 * 1024 + colH) * 1152 + (lhalf ? 16 : 0);
    const __bf16* Brow2 = WtL + (size_t)(2 * 1024 + colH) * 1152 + (lhalf ? 16 : 0);
    const __bf16* Brow3 = WtL + (size_t)(3 * 1024 + colH) * 1152 + (lhalf ? 16 : 0);

    v8f acc0 = {}, acc1 = {}, acc2 = {}, acc3 = {};
    for (int kt = 0; kt < 36; ++kt) {
        v16bf a = load_afrag(Arow, kt);
        acc0 = wmma_mac_a(a, Brow0, kt, acc0);   // i gate
        acc1 = wmma_mac_a(a, Brow1, kt, acc1);   // f gate
        acc2 = wmma_mac_a(a, Brow2, kt, acc2);   // g gate
        acc3 = wmma_mac_a(a, Brow3, kt, acc3);   // o gate
    }

    const float bi = b_lstm[0 * 1024 + colH];
    const float bf = b_lstm[1 * 1024 + colH];
    const float bg = b_lstm[2 * 1024 + colH];
    const float bo = b_lstm[3 * 1024 + colH];

#pragma unroll
    for (int r = 0; r < 8; ++r) {
        // C/D layout: VGPR r -> (M = r + 8*lhalf, N = lmod); same for all accs
        int row = rt * 16 + r + (lhalf << 3);    // batch index
        float ig = acc0[r] + bi;
        float fg = acc1[r] + bf;
        float gg = acc2[r] + bg;
        float og = acc3[r] + bo;
        float cold = c[row * 1024 + colH];
        float cn = sigf(fg) * cold + sigf(ig) * tanhf(gg);
        float hn = sigf(og) * tanhf(cn);
        c[row * 1024 + colH] = cn;
        __bf16 hb = (__bf16)hn;
        hbf[row * 1024 + colH] = hb;
        hr[row * 1088 + colH]  = hb;
    }
}

// ---------------------------------------------------------------------------
// Head GEMM: p = h @ W_head + b_head.  grid = (272/16=17, 4), block = 32.
// ---------------------------------------------------------------------------
__global__ void k_head(const __bf16* __restrict__ hbf,
                       const __bf16* __restrict__ WtH,
                       const float* __restrict__ b_head,
                       float* __restrict__ pout) {
    const int lane = threadIdx.x & 31;
    const int ct = blockIdx.x, rt = blockIdx.y;
    const int lhalf = lane >> 4, lmod = lane & 15;
    const int col = ct * 16 + lmod;
    const __bf16* Arow = hbf + (size_t)(rt * 16 + lmod) * 1024 + (lhalf ? 16 : 0);
    const __bf16* Brow = WtH + (size_t)col * 1024 + (lhalf ? 16 : 0);

    v8f acc = {};
    for (int kt = 0; kt < 32; ++kt)
        acc = wmma_mac(Arow, Brow, kt, acc);

    const float bias = (col < 268) ? b_head[col] : 0.0f;
#pragma unroll
    for (int r = 0; r < 8; ++r) {
        int row = rt * 16 + r + (lhalf << 3);
        pout[row * 272 + col] = acc[r] + bias;
    }
}

// ---------------------------------------------------------------------------
// NTM addressing + memory update + reads.  grid = B, block = 128.
// ---------------------------------------------------------------------------
__device__ __forceinline__ float blk_max(float* red, int tid, float v) {
    red[tid] = v; __syncthreads();
    for (int s = 64; s > 0; s >>= 1) {
        if (tid < s) red[tid] = fmaxf(red[tid], red[tid + s]);
        __syncthreads();
    }
    float r = red[0]; __syncthreads(); return r;
}
__device__ __forceinline__ float blk_sum(float* red, int tid, float v) {
    red[tid] = v; __syncthreads();
    for (int s = 64; s > 0; s >>= 1) {
        if (tid < s) red[tid] += red[tid + s];
        __syncthreads();
    }
    float r = red[0]; __syncthreads(); return r;
}

__global__ void k_addr(const float* __restrict__ p,
                       float* __restrict__ M,
                       float* __restrict__ wprev,
                       __bf16* __restrict__ rbf,
                       __bf16* __restrict__ hr) {
    const int b = blockIdx.x;
    const int tid = threadIdx.x;

    __shared__ float sk[2][64];
    __shared__ float serase[64], sadd[64];
    __shared__ float sbeta[2], sgate[2], sshift[2][3], sgamma[2], skn[2];
    __shared__ float swg[2][128];
    __shared__ float sw[2][128];
    __shared__ float red[128];

    const float* pb = p + b * 272;
    { int h = tid >> 6, j = tid & 63; sk[h][j] = tanhf(pb[h * 70 + j]); }
    if (tid < 64) {
        serase[tid] = sigf(pb[140 + tid]);
        sadd[tid]   = tanhf(pb[140 + 64 + tid]);
    }
    if (tid < 2) {
        const float* hp = pb + tid * 70;
        sbeta[tid]  = softplusf(hp[64]);
        sgate[tid]  = sigf(hp[65]);
        float s0 = hp[66], s1 = hp[67], s2 = hp[68];
        float mx = fmaxf(s0, fmaxf(s1, s2));
        float e0 = expf(s0 - mx), e1 = expf(s1 - mx), e2 = expf(s2 - mx);
        float es = e0 + e1 + e2;
        sshift[tid][0] = e0 / es; sshift[tid][1] = e1 / es; sshift[tid][2] = e2 / es;
        sgamma[tid] = 1.0f + softplusf(hp[69]);
    }
    __syncthreads();
    if (tid < 2) {
        float s = 0.0f;
        for (int j = 0; j < 64; ++j) s += sk[tid][j] * sk[tid][j];
        skn[tid] = sqrtf(s) + EPSC;
    }
    __syncthreads();

    // --- per memory-row phase: cosine sim -> content weights -> sharpen ----
    const int m = tid;                               // 128 rows
    const float* Mrow = M + (size_t)(b * 128 + m) * 64;
    float nrm = 0.0f, d0 = 0.0f, d1 = 0.0f;
    for (int n = 0; n < 64; ++n) {
        float Mv = Mrow[n];
        nrm += Mv * Mv;
        d0  += Mv * sk[0][n];
        d1  += Mv * sk[1][n];
    }
    nrm = sqrtf(nrm) + EPSC;
    float sim[2] = { d0 / (nrm * skn[0]), d1 / (nrm * skn[1]) };

    for (int h = 0; h < 2; ++h) {
        float a  = sbeta[h] * sim[h];
        float mx = blk_max(red, tid, a);
        float ex = expf(a - mx);
        float sm = blk_sum(red, tid, ex);
        float wc = ex / sm;
        float g  = sgate[h];
        swg[h][m] = g * wc + (1.0f - g) * wprev[b * 256 + h * 128 + m];
    }
    __syncthreads();
    for (int h = 0; h < 2; ++h) {
        float ws = sshift[h][0] * swg[h][(m + 1) & 127]
                 + sshift[h][1] * swg[h][m]
                 + sshift[h][2] * swg[h][(m + 127) & 127];
        float wp = powf(ws + EPSC, sgamma[h]);
        float sm = blk_sum(red, tid, wp);
        float wv = wp / sm;
        sw[h][m] = wv;
        wprev[b * 256 + h * 128 + m] = wv;
    }
    __syncthreads();

    // --- per-column phase: reads (old M) then erase/add update ------------
    if (tid < 64) {
        const int n = tid;
        const float er = serase[n], ad = sadd[n];
        float rsum = 0.0f;
        for (int mm = 0; mm < 128; ++mm) {
            float* Mp = M + (size_t)(b * 128 + mm) * 64 + n;
            float Mv = *Mp;
            rsum += sw[0][mm] * Mv;                              // read head
            *Mp = Mv * (1.0f - sw[1][mm] * er) + sw[1][mm] * ad; // write head
        }
        __bf16 rb = (__bf16)rsum;
        rbf[b * 64 + n] = rb;
        hr[b * 1088 + 1024 + n] = rb;
    }
}

// ---------------------------------------------------------------------------
// Output GEMM: sigmoid([h,r] @ W_out + b_out).  grid = (4, 4), block = 32.
// Only launched for the second scan (t >= 64).
// ---------------------------------------------------------------------------
__global__ void k_out(const __bf16* __restrict__ hr,
                      const __bf16* __restrict__ WtO,
                      const float* __restrict__ b_out,
                      float* __restrict__ out, int tt) {
    const int lane = threadIdx.x & 31;
    const int ct = blockIdx.x, rt = blockIdx.y;
    const int lhalf = lane >> 4, lmod = lane & 15;
    const int col = ct * 16 + lmod;
    const __bf16* Arow = hr  + (size_t)(rt * 16 + lmod) * 1088 + (lhalf ? 16 : 0);
    const __bf16* Brow = WtO + (size_t)col * 1088 + (lhalf ? 16 : 0);

    v8f acc = {};
    for (int kt = 0; kt < 34; ++kt)
        acc = wmma_mac(Arow, Brow, kt, acc);

    const float bias = b_out[col];
#pragma unroll
    for (int r = 0; r < 8; ++r) {
        int row = rt * 16 + r + (lhalf << 3);                 // batch index
        out[(size_t)row * 62 * 64 + tt * 64 + col] = sigf(acc[r] + bias);
    }
}

// ---------------------------------------------------------------------------
// Host side
// ---------------------------------------------------------------------------
extern "C" void kernel_launch(void* const* d_in, const int* in_sizes, int n_in,
                              void* d_out, int out_size, void* d_ws, size_t ws_size,
                              hipStream_t stream) {
    const float* x      = (const float*)d_in[0];   // (64, 64, 64)
    const float* W_lstm = (const float*)d_in[1];   // (1152, 4096)
    const float* b_lstm = (const float*)d_in[2];   // (4096,)
    const float* W_head = (const float*)d_in[3];   // (1024, 268)
    const float* b_head = (const float*)d_in[4];   // (268,)
    const float* W_out  = (const float*)d_in[5];   // (1088, 64)
    const float* b_out  = (const float*)d_in[6];   // (64,)
    float* out = (float*)d_out;                    // (64, 62, 64)

    char* ws = (char*)d_ws;
    size_t off = 0;
    auto carve = [&](size_t bytes) -> char* {
        char* p = ws + off;
        off += (bytes + 255) & ~(size_t)255;
        return p;
    };
    __bf16* WtL   = (__bf16*)carve((size_t)4096 * 1152 * 2);  // W_lstm^T bf16
    __bf16* WtH   = (__bf16*)carve((size_t)272  * 1024 * 2);  // W_head^T bf16 (padded)
    __bf16* WtO   = (__bf16*)carve((size_t)64   * 1088 * 2);  // W_out^T bf16
    __bf16* A     = (__bf16*)carve((size_t)64 * 1152 * 2);    // [x,r,h] activations
    __bf16* hbf   = (__bf16*)carve((size_t)64 * 1024 * 2);    // h bf16
    __bf16* hr    = (__bf16*)carve((size_t)64 * 1088 * 2);    // [h,r] bf16
    __bf16* rbf   = (__bf16*)carve((size_t)64 * 64 * 2);      // reads bf16
    float*  cbuf  = (float*) carve((size_t)64 * 1024 * 4);    // LSTM cell state
    float*  pbuf  = (float*) carve((size_t)64 * 272 * 4);     // head output
    float*  Mbuf  = (float*) carve((size_t)64 * 128 * 64 * 4);// NTM memory
    float*  wpbuf = (float*) carve((size_t)64 * 256 * 4);     // prev weights (B,2,128)

    // ---- one-time prep (per call): weight transpose/convert + state init ---
    {
        int n;
        n = 4096 * 1152; k_cvt_t<<<(n + 255) / 256, 256, 0, stream>>>(W_lstm, WtL, 1152, 4096, 4096);
        n = 272 * 1024;  k_cvt_t<<<(n + 255) / 256, 256, 0, stream>>>(W_head, WtH, 1024, 268, 272);
        n = 64 * 1088;   k_cvt_t<<<(n + 255) / 256, 256, 0, stream>>>(W_out,  WtO, 1088, 64, 64);

        n = 64 * 1024;       k_fill_f32<<<(n + 255) / 256, 256, 0, stream>>>(cbuf, 0.0f, n);
        n = 64 * 128 * 64;   k_fill_f32<<<(n + 255) / 256, 256, 0, stream>>>(Mbuf, 1e-6f, n);
        n = 64 * 1024;       k_fill_bf16<<<(n + 255) / 256, 256, 0, stream>>>(hbf, n);
        n = 64 * 64;         k_fill_bf16<<<(n + 255) / 256, 256, 0, stream>>>(rbf, n);
        n = 64 * 256;        k_init_wprev<<<(n + 255) / 256, 256, 0, stream>>>(wpbuf, n);
    }

    // ---- recurrence: 64 input steps + 62 zero-input steps ------------------
    for (int t = 0; t < 126; ++t) {
        k_pack<<<64, 128, 0, stream>>>(A, x, rbf, hbf, t);
        k_lstm<<<dim3(64, 4), 32, 0, stream>>>(A, WtL, b_lstm, cbuf, hbf, hr);
        k_head<<<dim3(17, 4), 32, 0, stream>>>(hbf, WtH, b_head, pbuf);
        k_addr<<<64, 128, 0, stream>>>(pbuf, Mbuf, wpbuf, rbf, hr);
        if (t >= 64)
            k_out<<<dim3(4, 4), 32, 0, stream>>>(hr, WtO, b_out, out, t - 64);
    }
}
